// ScaledDotProductAttention_1786706395426
// MI455X (gfx1250) — matrix operand order repaired and checked by reading, not compiled, not run on hardware
//
#include <hip/hip_runtime.h>
#include <hip/hip_bf16.h>

typedef __attribute__((ext_vector_type(16))) __bf16 v16bf;
typedef __attribute__((ext_vector_type(8)))  float  v8f;

#define BATCH 16
#define LSEQ  2048
#define DDIM  128
#define QT    16            // q rows per workgroup
#define NWAVE 8             // wave32s per workgroup
#define SCALE 0.08838834764831845f   // 1/sqrt(128)
#define VSTRIDE 20          // padded row stride (floats) for V staging, 16B aligned

// LDS: sc[16][2048] fp32 (128KB) + rowmax[16] + rowsum[16]
//      + per-wave V staging: [NWAVE][2 bufs][32 rows][VSTRIDE floats]
#define SMEM_FLOATS (QT * LSEQ + 2 * QT + NWAVE * 2 * 32 * VSTRIDE)

// Async copy of one contiguous 64B row (global -> LDS), 4x B128.
// INST_OFFSET applies to both global and LDS addresses (ISA 08 §4.4).
__device__ __forceinline__ void async_copy_row64(const float* g, unsigned lds_off) {
    unsigned long long ga = (unsigned long long)(uintptr_t)g;
    asm volatile("global_load_async_to_lds_b128 %0, %1, off"           :: "v"(lds_off), "v"(ga) : "memory");
    asm volatile("global_load_async_to_lds_b128 %0, %1, off offset:16" :: "v"(lds_off), "v"(ga) : "memory");
    asm volatile("global_load_async_to_lds_b128 %0, %1, off offset:32" :: "v"(lds_off), "v"(ga) : "memory");
    asm volatile("global_load_async_to_lds_b128 %0, %1, off offset:48" :: "v"(lds_off), "v"(ga) : "memory");
}

__global__ __launch_bounds__(256, 1)
void attn_fused_kernel(const float* __restrict__ Q, const float* __restrict__ K,
                       const float* __restrict__ V, const int* __restrict__ lens,
                       float* __restrict__ Out, float* __restrict__ P)
{
    extern __shared__ float smem[];
    float* sc     = smem;                 // [QT][LSEQ]
    float* rowmax = smem + QT * LSEQ;     // [QT]
    float* rowsum = rowmax + QT;          // [QT]
    float* vstall = rowsum + QT;          // V staging base

    const int tid  = threadIdx.x;
    const int wave = tid >> 5;
    const int lane = tid & 31;
    const int col  = lane & 15;           // N index within 16x16 tile
    const int hi   = lane >> 4;           // lane half (ISA K-split)

    const int blk   = blockIdx.x;
    const int b     = blk >> 7;           // / (LSEQ/QT)
    const int qtile = blk & 127;
    const int qbase = qtile * QT;
    const int len   = lens[b];

    const float* Qb = Q + (size_t)b * LSEQ * DDIM;
    const float* Kb = K + (size_t)b * LSEQ * DDIM;
    const float* Vb = V + (size_t)b * LSEQ * DDIM;

    // ---------------- A fragments: Q tile, pre-scaled by 1/sqrt(D) ----------
    // A 16x32 bf16 layout: lane holds row M=col; VGPR v<4 -> K = 2v + 8*hi,
    // VGPR v>=4 -> K = 16 + 2(v-4) + 8*hi (pairs in low/high halves).
    v16bf aq[4];
    {
        const float* qrow = Qb + (size_t)(qbase + col) * DDIM;
        #pragma unroll
        for (int c = 0; c < 4; ++c) {
            const int dbase = 32 * c + 8 * hi;
            #pragma unroll
            for (int v = 0; v < 8; ++v) {
                const int d = dbase + ((v < 4) ? (2 * v) : (16 + 2 * (v - 4)));
                aq[c][2 * v]     = (__bf16)(qrow[d]     * SCALE);
                aq[c][2 * v + 1] = (__bf16)(qrow[d + 1] * SCALE);
            }
        }
    }

    // ---------------- Phase 1: S = (Q/T) K^T, masked, to LDS ----------------
    for (int i = 0; i < (LSEQ / 16) / NWAVE; ++i) {
        const int kt  = wave + i * NWAVE;
        const int kb0 = kt * 16;
        const float* krow = Kb + (size_t)(kb0 + col) * DDIM;
        if (i + 1 < (LSEQ / 16) / NWAVE)
            __builtin_prefetch(Kb + (size_t)(kb0 + NWAVE * 16 + col) * DDIM, 0, 0);

        v8f acc = {};
        #pragma unroll
        for (int c = 0; c < 4; ++c) {
            // B 32x16 bf16 layout: lane holds column n=col; VGPR v<4 -> K = 2v + 8*hi, v>=4 -> K = 16 + 2(v-4) + 8*hi.
            v16bf bk;
            const int dbase = 32 * c + 8 * hi;
            #pragma unroll
            for (int v = 0; v < 8; ++v) {
                const int d = dbase + ((v < 4) ? (2 * v) : (16 + 2 * (v - 4)));
                bk[2 * v]     = (__bf16)krow[d];
                bk[2 * v + 1] = (__bf16)krow[d + 1];
            }
            acc = __builtin_amdgcn_wmma_f32_16x16x32_bf16(
                      false, aq[c], false, bk, (short)0, acc, false, false);
        }
        const bool valid = (kb0 + col) < len;
        #pragma unroll
        for (int r = 0; r < 8; ++r) {
            const int m = r + 8 * hi;           // C/D layout: row = vgpr + 8*hi
            sc[m * LSEQ + kb0 + col] = valid ? acc[r] : -__builtin_inff();
        }
    }
    __syncthreads();

    // ---------------- Phase 2: exact softmax row max / sum ------------------
    {
        const int m = tid >> 4, l16 = tid & 15;
        float mx = -__builtin_inff();
        for (int k = l16; k < LSEQ; k += 16) mx = fmaxf(mx, sc[m * LSEQ + k]);
        #pragma unroll
        for (int off = 8; off >= 1; off >>= 1) mx = fmaxf(mx, __shfl_xor(mx, off, 16));
        if (l16 == 0) rowmax[m] = mx;
    }
    __syncthreads();
    {
        const int m = tid >> 4, l16 = tid & 15;
        const float mx = rowmax[m];
        float sum = 0.f;
        for (int k = l16; k < LSEQ; k += 16) {
            const float s = sc[m * LSEQ + k];
            sum += (s == -__builtin_inff()) ? 0.f : __expf(s - mx);
        }
        #pragma unroll
        for (int off = 8; off >= 1; off >>= 1) sum += __shfl_xor(sum, off, 16);
        if (l16 == 0) rowsum[m] = sum;
    }
    __syncthreads();

    // ------- Phase 3: normalize; stream p to HBM (NT); keep p in LDS --------
    {
        float* Prow = P + (size_t)b * LSEQ * LSEQ + (size_t)qbase * LSEQ;
        const int m = tid >> 4, l16 = tid & 15;
        const float mx  = rowmax[m];
        const float s_  = rowsum[m];
        const float inv = (s_ > 0.f) ? (1.f / s_) : 0.f;  // fully-masked row -> 0
        for (int k = l16; k < LSEQ; k += 16) {
            const float s  = sc[m * LSEQ + k];
            const float pv = (s == -__builtin_inff()) ? 0.f : __expf(s - mx) * inv;
            sc[m * LSEQ + k] = pv;                               // reuse for PV
            __builtin_nontemporal_store(pv, &Prow[(size_t)m * LSEQ + k]);
        }
    }
    __syncthreads();

    // ---------------- Phase 4: Out = P (16x2048) x V (2048x128) -------------
    // Wave w owns d-columns [16w, 16w+16). Its 32x16 V slice per k-chunk is
    // staged into a private double-buffered LDS region via async-to-LDS copies
    // (one 64B row per lane per buffer), overlapped with WMMA via ASYNCcnt.
    {
        const int d0 = wave * 16;
        float* vst = vstall + wave * (2 * 32 * VSTRIDE);
        const float* gbase = Vb + d0;

        // prologue: stage chunk 0 into buffer 0
        async_copy_row64(gbase + (size_t)lane * DDIM,
                         (unsigned)(uintptr_t)(const void*)(vst + lane * VSTRIDE));

        v8f oacc = {};
        int buf = 0;
        for (int c = 0; c < LSEQ / 32; ++c) {
            if (c + 1 < LSEQ / 32) {
                async_copy_row64(gbase + (size_t)((c + 1) * 32 + lane) * DDIM,
                                 (unsigned)(uintptr_t)(const void*)
                                     (vst + (buf ^ 1) * (32 * VSTRIDE) + lane * VSTRIDE));
                asm volatile("s_wait_asynccnt 0x4" ::: "memory"); // chunk c resident
            } else {
                asm volatile("s_wait_asynccnt 0x0" ::: "memory");
            }

            const int kb = c * 32;
            // A fragment from LDS probabilities (row m = col).
            v16bf ap;
            {
                const float* prow = sc + col * LSEQ + kb + 8 * hi;
                #pragma unroll
                for (int v = 0; v < 8; ++v) {
                    const int kk = (v < 4) ? (2 * v) : (16 + 2 * (v - 4));
                    ap[2 * v]     = (__bf16)prow[kk];
                    ap[2 * v + 1] = (__bf16)prow[kk + 1];
                }
            }
            // B fragment from staged V slice: row = (v<4 ? 2v : 16+2(v-4)) + 8*hi, column = col.
            v16bf bv;
            {
                const float* vb0 = vst + buf * (32 * VSTRIDE);
                #pragma unroll
                for (int v = 0; v < 8; ++v) {
                    const int r0 = 8 * hi + ((v < 4) ? (2 * v) : (16 + 2 * (v - 4)));
                    bv[2 * v]     = (__bf16)vb0[r0 * VSTRIDE + col];
                    bv[2 * v + 1] = (__bf16)vb0[(r0 + 1) * VSTRIDE + col];
                }
            }
            oacc = __builtin_amdgcn_wmma_f32_16x16x32_bf16(
                       false, ap, false, bv, (short)0, oacc, false, false);
            buf ^= 1;
        }

        float* Ob = Out + (size_t)b * LSEQ * DDIM + (size_t)qbase * DDIM;
        #pragma unroll
        for (int r = 0; r < 8; ++r) {
            const int m = r + 8 * hi;
            __builtin_nontemporal_store(oacc[r], &Ob[(size_t)m * DDIM + d0 + col]);
        }
    }
}

extern "C" void kernel_launch(void* const* d_in, const int* in_sizes, int n_in,
                              void* d_out, int out_size, void* d_ws, size_t ws_size,
                              hipStream_t stream) {
    (void)in_sizes; (void)n_in; (void)d_ws; (void)ws_size; (void)out_size;
    const float* q    = (const float*)d_in[0];
    const float* k    = (const float*)d_in[1];
    const float* v    = (const float*)d_in[2];
    const int*   lens = (const int*)d_in[3];

    float* out = (float*)d_out;                                   // [B,L,D]
    float* p   = (float*)d_out + (size_t)BATCH * LSEQ * DDIM;     // [B,L,L]

    const int    blocks = BATCH * (LSEQ / QT);                    // 2048
    const size_t smem   = (size_t)SMEM_FLOATS * sizeof(float);    // ~168KB

    attn_fused_kernel<<<blocks, 256, smem, stream>>>(q, k, v, lens, out, p);
}
